// MaskedSelfAttention_44272522887253
// MI455X (gfx1250) — matrix operand order repaired and checked by reading, not compiled, not run on hardware
//
#include <hip/hip_runtime.h>
#include <hip/hip_bf16.h>

// ---- problem constants (match reference) ----
constexpr int B_ = 2, T_ = 2048, C_ = 1024;
constexpr int H_ = 16, G_ = 4, D_ = 64;
constexpr int GROUP_ = H_ / G_;          // 4
constexpr int WIN_ = 512, SINK_ = 4;
constexpr float SCALE_ = 0.125f;         // D^-0.5
constexpr float BASE_ = 10000.0f;

typedef __attribute__((ext_vector_type(16))) _Float16 v16h;
typedef __attribute__((ext_vector_type(8)))  _Float16 v8h;
typedef __attribute__((ext_vector_type(8)))  float    v8f;
typedef __attribute__((ext_vector_type(4)))  unsigned u32x4;
typedef __attribute__((ext_vector_type(8)))  int      i32x8;
typedef __attribute__((ext_vector_type(4)))  int      i32x4;

static __device__ __forceinline__ v16h mk16(v8h lo, v8h hi) {
  return __builtin_shufflevector(lo, hi, 0,1,2,3,4,5,6,7,8,9,10,11,12,13,14,15);
}

// async 16B copy global -> LDS (GV mode, saddr = off); tracked by ASYNCcnt
static __device__ __forceinline__ void async_copy_b128(const _Float16* gp, _Float16* lp) {
  unsigned lds_addr = (unsigned)(unsigned long long)lp;   // low 32 bits = LDS offset
  asm volatile("global_load_async_to_lds_b128 %0, %1, off"
               :: "v"(lds_addr), "v"(gp) : "memory");
}
static __device__ __forceinline__ void async_wait0() {
  asm volatile("s_wait_asynccnt 0x0" ::: "memory");
}

// TDM: load a tile_rows x 32-elem f16 tile from a row-major [tensor_rows x K] matrix
// starting at (row0, k0) into LDS at lds_addr, padding each 64B row to an 80B
// (BKP=40 f16) LDS stride via D# pad fields. Tracked by TENSORcnt.
static __device__ __forceinline__ void tdm_load_tile(
    const _Float16* gbase, unsigned lds_addr,
    int tensor_rows, int K, int row0, int k0, int tile_rows)
{
  unsigned long long ga = (unsigned long long)gbase
                        + ((unsigned long long)(unsigned)row0 * (unsigned)K
                           + (unsigned)k0) * 2ull;
  u32x4 g0;
  g0[0] = 1u;                                             // count=1, user descriptor
  g0[1] = lds_addr;                                       // LDS byte address
  g0[2] = (unsigned)ga;                                   // global_addr[31:0]
  g0[3] = (unsigned)((ga >> 32) & 0x01FFFFFFull) | (2u << 30);  // addr[56:32] | type=2

  i32x8 g1;
  // data_size=2B (1), pad_enable, pad_interval=3 (16 DWORDs), pad_amount=3 (4 DWORDs)
  g1[0] = (1 << 16) | (1 << 20) | (3 << 22) | (3 << 25);
  g1[1] = (K & 0xFFFF) << 16;                             // tensor_dim0[15:0]
  g1[2] = ((K >> 16) & 0xFFFF) | ((tensor_rows & 0xFFFF) << 16); // dim0 hi | dim1 lo
  g1[3] = ((tensor_rows >> 16) & 0xFFFF) | (32 << 16);    // dim1 hi | tile_dim0=32
  g1[4] = tile_rows & 0xFFFF;                             // tile_dim1 (tile_dim2=0)
  g1[5] = K;                                              // tensor_dim0_stride[31:0]
  g1[6] = 0;
  g1[7] = 0;

  i32x4 gz4 = {0, 0, 0, 0};
  i32x8 gz8 = {0, 0, 0, 0, 0, 0, 0, 0};
  // clang-23 / amdgpu-toolchain 6-arg form
  __builtin_amdgcn_tensor_load_to_lds(g0, g1, gz4, gz4, gz8, 0);
}

// ---------------- conversion / packing kernels ----------------
__global__ void cvt_f32_f16(const float* __restrict__ src, _Float16* __restrict__ dst, int n) {
  int i = blockIdx.x * blockDim.x + threadIdx.x;
  if (i < n) dst[i] = (_Float16)src[i];
}

// dst[n*K + k] = src[k*N + n]  (f32 -> f16 transpose)
__global__ void cvt_transpose(const float* __restrict__ src, _Float16* __restrict__ dst,
                              int N, int K) {
  int i = blockIdx.x * blockDim.x + threadIdx.x;
  if (i >= N * K) return;
  int n = i / K, k = i % K;
  dst[i] = (_Float16)src[(size_t)k * N + n];
}

// WhT [1536][1024]: concatenated, transposed Wq|Wk|Wv in f16
__global__ void pack_wqkv_T(const float* __restrict__ Wq, const float* __restrict__ Wk,
                            const float* __restrict__ Wv, _Float16* __restrict__ WhT) {
  int i = blockIdx.x * blockDim.x + threadIdx.x;           // over 1536*1024
  if (i >= 1536 * 1024) return;
  int n = i >> 10;          // output row (QKV column) 0..1535
  int k = i & 1023;         // reduction index
  float v;
  if (n < 1024)      v = Wq[(size_t)k * 1024 + n];
  else if (n < 1280) v = Wk[(size_t)k * 256 + (n - 1024)];
  else               v = Wv[(size_t)k * 256 + (n - 1280)];
  WhT[i] = (_Float16)v;
}

__global__ void pack_bias(const float* __restrict__ bq, const float* __restrict__ bk,
                          const float* __restrict__ bv, float* __restrict__ bias) {
  int i = blockIdx.x * blockDim.x + threadIdx.x;
  if (i < 1024) bias[i] = bq[i];
  else if (i < 1280) bias[i] = bk[i - 1024];
  else if (i < 1536) bias[i] = bv[i - 1280];
}

// ---------------- generic f16xf16 -> f32 tiled WMMA GEMM ----------------
// C[M,N] = A[M,K] * BT[N,K]^T + bias[N];  M%64==0, N%128==0, K%32==0
// A tile staged by the Tensor Data Mover (wave 0 issues one descriptor),
// B tile staged with per-lane async b128 copies.
constexpr int BM = 64, BN = 128, BK = 32, BKP = 40; // BKP: padded LDS row (16B-aligned stride)

__global__ __launch_bounds__(256) void gemm_f16(
    const _Float16* __restrict__ A, const _Float16* __restrict__ BT,
    const float* __restrict__ bias, float* __restrict__ C,
    int M, int N, int K)
{
  __shared__ alignas(16) _Float16 As[BM * BKP];
  __shared__ alignas(16) _Float16 Bs[BN * BKP];

  const int tid  = threadIdx.x;
  const int lane = tid & 31;
  const int wave = tid >> 5;
  const int wm = wave & 3;      // 4 M-subtiles of 16
  const int wn = wave >> 2;     // 2 N-subtiles of 64
  const int m0 = blockIdx.x * BM;
  const int n0 = blockIdx.y * BN;
  const int rrow = lane & 15;
  const unsigned ldsA = (unsigned)(unsigned long long)&As[0];

  v8f acc[4];
  #pragma unroll
  for (int nt = 0; nt < 4; ++nt)
    #pragma unroll
    for (int r = 0; r < 8; ++r) acc[nt][r] = 0.f;

  // per-thread B staging assignments (16B chunks): 128x32 = 512 chunks, 2/thread
  const int brow0 = (tid * 2) >> 2, bch0 = (tid * 2) & 3;
  const int brow1 = (tid * 2 + 1) >> 2, bch1 = (tid * 2 + 1) & 3;

  for (int k0 = 0; k0 < K; k0 += BK) {
    async_copy_b128(BT + (size_t)(n0 + brow0) * K + k0 + bch0 * 8, &Bs[brow0 * BKP + bch0 * 8]);
    async_copy_b128(BT + (size_t)(n0 + brow1) * K + k0 + bch1 * 8, &Bs[brow1 * BKP + bch1 * 8]);
    if (wave == 0) {
      tdm_load_tile(A, ldsA, M, K, m0, k0, BM);           // 64x32 tile, padded to BKP
      __builtin_amdgcn_s_wait_tensorcnt(0);
    }
    async_wait0();
    __syncthreads();

    // A fragment: 16x32 f16, contiguous k layout (lanes 0-15: k 0-15, lanes 16-31: k 16-31)
    int row = wm * 16 + rrow;
    int kb  = (lane < 16) ? 0 : 16;
    v8h alo = *(const v8h*)(&As[row * BKP + kb]);
    v8h ahi = *(const v8h*)(&As[row * BKP + kb + 8]);
    v16h afrag = mk16(alo, ahi);

    int kb2 = (lane < 16) ? 0 : 16;
    #pragma unroll
    for (int nt = 0; nt < 4; ++nt) {
      int col = wn * 64 + nt * 16 + rrow;
      v8h blo = *(const v8h*)(&Bs[col * BKP + kb2]);
      v8h bhi = *(const v8h*)(&Bs[col * BKP + kb2 + 8]);
      v16h bfrag = mk16(blo, bhi);
      acc[nt] = __builtin_amdgcn_wmma_f32_16x16x32_f16(
          false, afrag, false, bfrag, (short)0, acc[nt], false, false);
    }
    __syncthreads();
  }

  #pragma unroll
  for (int nt = 0; nt < 4; ++nt) {
    int col = n0 + wn * 64 + nt * 16 + rrow;
    float bvv = bias ? bias[col] : 0.f;
    #pragma unroll
    for (int r = 0; r < 8; ++r) {
      int mrow = m0 + wm * 16 + r + 8 * (lane >> 4);
      C[(size_t)mrow * N + col] = acc[nt][r] + bvv;
    }
  }
}

// ---------------- RoPE + head-layout packing ----------------
// qkv: [B*T, 1536] f32 -> qh [B,H,T,D] f16 (roped), kh [B,G,T,D] f16 (roped),
//                         vT [B,G,D,T] f16 (transposed for P*V B-operand)
__global__ void rope_pack(const float* __restrict__ qkv, const int* __restrict__ posp,
                          _Float16* __restrict__ qh, _Float16* __restrict__ kh,
                          _Float16* __restrict__ vT)
{
  int idx = blockIdx.x * blockDim.x + threadIdx.x;
  const int SLOTS = 768;                       // 512 q-pairs + 128 k-pairs + 128 v-pairs
  if (idx >= B_ * T_ * SLOTS) return;
  int slot = idx % SLOTS;
  int bt   = idx / SLOTS;
  int t = bt % T_, b = bt / T_;
  const float* row = qkv + (size_t)bt * 1536;
  float pos = (float)(*posp) + (float)t;

  if (slot < 512) {                            // Q
    int h = slot >> 5, d = (slot & 31) * 2;
    float xe = row[h * 64 + d], xo = row[h * 64 + d + 1];
    float th = pos * __powf(BASE_, -(float)d / (float)D_);
    float s, c; __sincosf(th, &s, &c);
    size_t base = ((size_t)(b * H_ + h) * T_ + t) * D_ + d;
    qh[base]     = (_Float16)(xe * c - xo * s);
    qh[base + 1] = (_Float16)(xe * s + xo * c);
  } else if (slot < 640) {                     // K
    int g = (slot - 512) >> 5, d = ((slot - 512) & 31) * 2;
    float xe = row[1024 + g * 64 + d], xo = row[1024 + g * 64 + d + 1];
    float th = pos * __powf(BASE_, -(float)d / (float)D_);
    float s, c; __sincosf(th, &s, &c);
    size_t base = ((size_t)(b * G_ + g) * T_ + t) * D_ + d;
    kh[base]     = (_Float16)(xe * c - xo * s);
    kh[base + 1] = (_Float16)(xe * s + xo * c);
  } else {                                     // V (no rope, transpose)
    int g = (slot - 640) >> 5, d = ((slot - 640) & 31) * 2;
    float v0 = row[1280 + g * 64 + d], v1 = row[1280 + g * 64 + d + 1];
    size_t base = ((size_t)(b * G_ + g) * D_ + d) * T_ + t;
    vT[base]      = (_Float16)v0;
    vT[base + T_] = (_Float16)v1;
  }
}

// ---------------- flash attention: 4 independent waves / block, 16 queries / wave ----------------
__global__ __launch_bounds__(128) void attn_kernel(
    const _Float16* __restrict__ qh, const _Float16* __restrict__ kh,
    const _Float16* __restrict__ vT, _Float16* __restrict__ outh)
{
  __shared__ alignas(16) _Float16 Pbuf[4 * 16 * 40];   // private slice per wave

  const int lane = threadIdx.x & 31;
  const int wav  = threadIdx.x >> 5;
  const int i0 = (blockIdx.x * 4 + wav) * 16;
  const int h = blockIdx.y, b = blockIdx.z;
  const int g = h / GROUP_;
  const int rrow = lane & 15;
  const int half = lane >> 4;
  _Float16* Pb = &Pbuf[wav * 16 * 40];

  const _Float16* Qp = qh + (size_t)(b * H_ + h) * T_ * D_;
  const _Float16* Kp = kh + (size_t)(b * G_ + g) * T_ * D_;
  const _Float16* Vp = vT + (size_t)(b * G_ + g) * D_ * T_;

  // Q fragments for the two 32-wide d-chunks (A-operand layout)
  v16h qfrag[2];
  {
    int kb = (lane < 16) ? 0 : 16;
    #pragma unroll
    for (int c = 0; c < 2; ++c) {
      v8h lo = *(const v8h*)(Qp + (size_t)(i0 + rrow) * D_ + c * 32 + kb);
      v8h hi = *(const v8h*)(Qp + (size_t)(i0 + rrow) * D_ + c * 32 + kb + 8);
      qfrag[c] = mk16(lo, hi);
    }
  }

  float m[8], l[8];
  v8f O[4];
  #pragma unroll
  for (int r = 0; r < 8; ++r) { m[r] = -1e30f; l[r] = 0.f; }
  #pragma unroll
  for (int dt = 0; dt < 4; ++dt)
    #pragma unroll
    for (int r = 0; r < 8; ++r) O[dt][r] = 0.f;

  const int kb2 = (lane < 16) ? 0 : 16;

  auto process = [&](int k0) {
    // ---- S = Q * K^T for 32 keys (two 16-key subtiles) ----
    v8f S[2];
    #pragma unroll
    for (int s = 0; s < 2; ++s)
      #pragma unroll
      for (int r = 0; r < 8; ++r) S[s][r] = 0.f;

    #pragma unroll
    for (int s = 0; s < 2; ++s) {
      int key = k0 + s * 16 + rrow;              // B-operand column = key index
      #pragma unroll
      for (int c = 0; c < 2; ++c) {              // d-chunks of 32
        v8h lo = *(const v8h*)(Kp + (size_t)key * D_ + c * 32 + kb2);
        v8h hi = *(const v8h*)(Kp + (size_t)key * D_ + c * 32 + kb2 + 8);
        v16h kf = mk16(lo, hi);
        S[s] = __builtin_amdgcn_wmma_f32_16x16x32_f16(
            false, qfrag[c], false, kf, (short)0, S[s], false, false);
      }
    }

    // ---- masking + online softmax ----
    #pragma unroll
    for (int r = 0; r < 8; ++r) {
      int irow = i0 + r + 8 * half;
      float s0 = S[0][r] * SCALE_, s1 = S[1][r] * SCALE_;
      int j0 = k0 + rrow, j1 = j0 + 16;
      bool v0 = (j0 <= irow) && ((j0 >= irow - WIN_) || (j0 < SINK_));
      bool v1 = (j1 <= irow) && ((j1 >= irow - WIN_) || (j1 < SINK_));
      s0 = v0 ? s0 : -1e30f;
      s1 = v1 ? s1 : -1e30f;
      float rm = fmaxf(s0, s1);
      #pragma unroll
      for (int off = 8; off; off >>= 1) rm = fmaxf(rm, __shfl_xor(rm, off, 16));
      float mn = fmaxf(m[r], rm);
      float alpha = __expf(m[r] - mn);
      float p0 = __expf(s0 - mn), p1 = __expf(s1 - mn);
      float rs = p0 + p1;
      #pragma unroll
      for (int off = 8; off; off >>= 1) rs += __shfl_xor(rs, off, 16);
      l[r] = l[r] * alpha + rs;
      m[r] = mn;
      #pragma unroll
      for (int dt = 0; dt < 4; ++dt) O[dt][r] *= alpha;
      // stash P tile (accumulator layout) to LDS for A-fragment reload
      Pb[(r + 8 * half) * 40 + rrow]      = (_Float16)p0;
      Pb[(r + 8 * half) * 40 + 16 + rrow] = (_Float16)p1;
    }
    asm volatile("s_wait_dscnt 0x0" ::: "memory");  // LDS is in-order per wave; keep compiler honest

    // ---- O += P * V ----
    int kbp = (lane < 16) ? 0 : 16;
    v8h plo = *(const v8h*)(&Pb[rrow * 40 + kbp]);
    v8h phi = *(const v8h*)(&Pb[rrow * 40 + kbp + 8]);
    v16h pf = mk16(plo, phi);
    #pragma unroll
    for (int dt = 0; dt < 4; ++dt) {
      int d = dt * 16 + rrow;                    // B-operand column = head-dim
      v8h vlo = *(const v8h*)(Vp + (size_t)d * T_ + k0 + kb2);
      v8h vhi = *(const v8h*)(Vp + (size_t)d * T_ + k0 + kb2 + 8);
      v16h vf = mk16(vlo, vhi);
      O[dt] = __builtin_amdgcn_wmma_f32_16x16x32_f16(
          false, pf, false, vf, (short)0, O[dt], false, false);
    }
    asm volatile("" ::: "memory");
  };

  // sink block first (keys 0..31; guarantees running max is finite for every row)
  process(0);
  // sliding-window blocks, 32-aligned, skipping block 0
  int wstart = (i0 - WIN_) & ~31;
  if (wstart < 32) wstart = 32;
  for (int k0 = wstart; k0 <= i0 + 15; k0 += 32) process(k0);

  // ---- normalize + store as f16 in [B*T, H*D] for the out-projection GEMM ----
  #pragma unroll
  for (int dt = 0; dt < 4; ++dt) {
    int d = dt * 16 + rrow;
    #pragma unroll
    for (int r = 0; r < 8; ++r) {
      int t = i0 + r + 8 * half;
      float val = O[dt][r] / l[r];
      outh[((size_t)(b * T_ + t) * H_ + h) * D_ + d] = (_Float16)val;
    }
  }
}

// ---------------- host-side launch ----------------
extern "C" void kernel_launch(void* const* d_in, const int* in_sizes, int n_in,
                              void* d_out, int out_size, void* d_ws, size_t ws_size,
                              hipStream_t stream) {
  const float* x  = (const float*)d_in[0];
  const float* Wq = (const float*)d_in[1];
  const float* bq = (const float*)d_in[2];
  const float* Wk = (const float*)d_in[3];
  const float* bk = (const float*)d_in[4];
  const float* Wv = (const float*)d_in[5];
  const float* bv = (const float*)d_in[6];
  const float* Wo = (const float*)d_in[7];
  const float* bo = (const float*)d_in[8];
  const int*  pos = (const int*)d_in[9];
  float* out = (float*)d_out;

  // workspace layout (bytes)
  char* w = (char*)d_ws;
  constexpr size_t SZ_XH    = (size_t)B_ * T_ * C_ * 2;            //  8 MB
  constexpr size_t SZ_WQKV  = (size_t)C_ * 1536 * 2;               //  3 MB
  constexpr size_t SZ_WO    = (size_t)C_ * C_ * 2;                 //  2 MB
  constexpr size_t SZ_BIAS  = 1536 * 4;
  constexpr size_t SZ_QKVF  = (size_t)B_ * T_ * 1536 * 4;         // 24 MB
  constexpr size_t SZ_QH    = (size_t)B_ * H_ * T_ * D_ * 2;      //  8 MB
  constexpr size_t SZ_KH    = (size_t)B_ * G_ * T_ * D_ * 2;      //  2 MB
  constexpr size_t SZ_VT    = SZ_KH;                               //  2 MB
  size_t off = 0;
  _Float16* xh     = (_Float16*)(w + off); off += SZ_XH;
  _Float16* wqkvT  = (_Float16*)(w + off); off += SZ_WQKV;
  _Float16* woT    = (_Float16*)(w + off); off += SZ_WO;
  float*    biasq  = (float*)   (w + off); off += ((SZ_BIAS + 255) & ~(size_t)255);
  float*    qkvf   = (float*)   (w + off); off += SZ_QKVF;
  _Float16* qh     = (_Float16*)(w + off); off += SZ_QH;
  _Float16* kh     = (_Float16*)(w + off); off += SZ_KH;
  _Float16* vT     = (_Float16*)(w + off); off += SZ_VT;
  _Float16* atth   = (_Float16*)(w + off); off += SZ_XH;
  (void)ws_size; (void)in_sizes; (void)n_in; (void)out_size;

  const int TPB = 256;
  const int nx = B_ * T_ * C_;
  hipLaunchKernelGGL(cvt_f32_f16, dim3((nx + TPB - 1) / TPB), dim3(TPB), 0, stream, x, xh, nx);
  const int nwqkv = 1536 * 1024;
  hipLaunchKernelGGL(pack_wqkv_T, dim3((nwqkv + TPB - 1) / TPB), dim3(TPB), 0, stream,
                     Wq, Wk, Wv, wqkvT);
  const int nwo = C_ * C_;
  hipLaunchKernelGGL(cvt_transpose, dim3((nwo + TPB - 1) / TPB), dim3(TPB), 0, stream,
                     Wo, woT, C_, C_);
  hipLaunchKernelGGL(pack_bias, dim3(6), dim3(TPB), 0, stream, bq, bk, bv, biasq);

  // QKV projection: [4096 x 1024] x [1024 x 1536]  (B passed transposed)
  hipLaunchKernelGGL(gemm_f16, dim3((B_ * T_) / BM, 1536 / BN), dim3(256), 0, stream,
                     xh, wqkvT, biasq, qkvf, B_ * T_, 1536, C_);

  const int nrp = B_ * T_ * 768;
  hipLaunchKernelGGL(rope_pack, dim3((nrp + TPB - 1) / TPB), dim3(TPB), 0, stream,
                     qkvf, pos, qh, kh, vT);

  hipLaunchKernelGGL(attn_kernel, dim3(T_ / 64, H_, B_), dim3(128), 0, stream,
                     qh, kh, vT, atth);

  // Out projection: [4096 x 1024] x [1024 x 1024] -> d_out  (B passed transposed)
  hipLaunchKernelGGL(gemm_f16, dim3((B_ * T_) / BM, C_ / BN), dim3(256), 0, stream,
                     atth, woT, bo, out, B_ * T_, C_, C_);
}
